// RNNLayer_34686155883191
// MI455X (gfx1250) — compile-verified, matrix operand order checked by reading, not run on hardware
//
#include <hip/hip_runtime.h>
#include <hip/hip_bf16.h>

// ---------------- problem constants ----------------
#define HH 2048   // hidden
#define TT 4096   // timesteps
#define EE 2048   // embed
#define CC 64     // chunk length (inner sequential steps)
#define NCK 64    // number of chunks = TT/CC
#define KT (HH/32)   // 64 K-tiles of 32
#define MT (HH/16)   // 128 M-tiles of 16

typedef __attribute__((ext_vector_type(16))) __bf16 v16bf;
typedef __attribute__((ext_vector_type(8)))  float  v8f;

union Bf16x16 { v16bf v; unsigned short u[16]; };

__device__ __forceinline__ unsigned short f2bf(float f) {
  unsigned int u = __float_as_uint(f);
  u += 0x7FFFu + ((u >> 16) & 1u);          // round-to-nearest-even
  return (unsigned short)(u >> 16);
}

__device__ __forceinline__ float bf2f(unsigned int ushort_bits) {
  return __uint_as_float(ushort_bits << 16);
}

// K index held by (lane, q) for both A(16x32) and B(32x16) bf16 WMMA fragments
__device__ __forceinline__ int kOf(int lane, int q) {
  int base = (lane < 16) ? 0 : 8;
  return (q < 8) ? (q + base) : (q + 8 + base);
}

// ---------------- wx_t = dot(inputs[t,:], W) ----------------
__global__ void wx_kernel(const float* __restrict__ inp,
                          const float* __restrict__ Wv,
                          float* __restrict__ wx) {
  int lane = threadIdx.x & 31;
  int row  = blockIdx.x * 8 + (threadIdx.x >> 5);
  float s = 0.f;
  for (int j = lane; j < EE; j += 32) s += inp[(size_t)row * EE + j] * Wv[j];
  for (int off = 16; off > 0; off >>= 1) s += __shfl_xor(s, off, 32);
  if (lane == 0) wx[row] = s;
}

// ---------------- pack f32 matrix -> bf16 A-fragment layout ----------------
__global__ void pack_a(const float* __restrict__ src, unsigned short* __restrict__ dst) {
  int gid = blockIdx.x * blockDim.x + threadIdx.x;
  int lane = gid & 31, tile = gid >> 5;
  if (tile >= MT * KT) return;
  int mt = tile / KT, kt = tile % KT;
  int M = mt * 16 + (lane & 15);
  Bf16x16 w;
#pragma unroll
  for (int q = 0; q < 16; q++) {
    int K = kt * 32 + kOf(lane, q);
    w.u[q] = f2bf(src[(size_t)M * HH + K]);
  }
  *(v16bf*)(dst + (size_t)tile * 512 + lane * 16) = w.v;
}

// ---------------- pack f32 matrix -> bf16 B-fragment layout (full 2048 cols) ---
__global__ void pack_b_full(const float* __restrict__ src, unsigned short* __restrict__ dst) {
  int gid = blockIdx.x * blockDim.x + threadIdx.x;
  int lane = gid & 31, tile = gid >> 5;
  if (tile >= KT * (HH / 16)) return;
  int kt = tile >> 7, ntg = tile & 127;
  int n = ntg * 16 + (lane & 15);
  Bf16x16 w;
#pragma unroll
  for (int q = 0; q < 16; q++) {
    int K = kt * 32 + kOf(lane, q);
    w.u[q] = f2bf(src[(size_t)K * HH + n]);
  }
  *(v16bf*)(dst + (size_t)tile * 512 + lane * 16) = w.v;
}

// ---------------- pack f32 matrix -> plain row-major bf16 ----------------
__global__ void pack_plain_bf16(const float* __restrict__ src, unsigned short* __restrict__ dst) {
  int gid = blockIdx.x * blockDim.x + threadIdx.x;  // 4 elems / thread
  size_t base = (size_t)gid * 4;
  if (base >= (size_t)HH * HH) return;
  float4 f = *(const float4*)(src + base);
  unsigned short o[4] = {f2bf(f.x), f2bf(f.y), f2bf(f.z), f2bf(f.w)};
  *(uint2*)(dst + base) = *(uint2*)o;
}

// ---------------- init packed-B state: col0=h0, others 0 ----------------
__global__ void init_pass1_b(const float* __restrict__ h0, unsigned short* __restrict__ dst) {
  int gid = blockIdx.x * blockDim.x + threadIdx.x;
  int lane = gid & 31, tile = gid >> 5;
  if (tile >= KT * 4) return;
  int kt = tile >> 2, nt = tile & 3;
  int n = nt * 16 + (lane & 15);
  Bf16x16 w;
#pragma unroll
  for (int q = 0; q < 16; q++) {
    int K = kt * 32 + kOf(lane, q);
    w.u[q] = (n == 0) ? f2bf(h0[K]) : (unsigned short)0;
  }
  *(v16bf*)(dst + (size_t)tile * 512 + lane * 16) = w.v;
}

// ---------------- init packed-B state: col0=h0, col n = estates[:,n-1] -------
__global__ void init_pass3_b(const float* __restrict__ h0, const float* __restrict__ est,
                             unsigned short* __restrict__ dst) {
  int gid = blockIdx.x * blockDim.x + threadIdx.x;
  int lane = gid & 31, tile = gid >> 5;
  if (tile >= KT * 4) return;
  int kt = tile >> 2, nt = tile & 3;
  int n = nt * 16 + (lane & 15);
  Bf16x16 w;
#pragma unroll
  for (int q = 0; q < 16; q++) {
    int K = kt * 32 + kOf(lane, q);
    float v = (n == 0) ? h0[K] : est[(size_t)K * NCK + (n - 1)];
    w.u[q] = f2bf(v);
  }
  *(v16bf*)(dst + (size_t)tile * 512 + lane * 16) = w.v;
}

// ---------------- main chunked step: G_new = U*G + wx  (WMMA bf16) ----------
// 256 threads = 8 waves. Waves 0..3 take K-tiles [0,32), waves 4..7 take [32,64)
// for the same four 16-row M-tiles; partial sums combined through two LDS slabs.
// K-loop is software-pipelined with two alternating register sets (no rotates).
// MODE 0: write plain f32 G to outp[i*NCK+n]   (pass 1: local sweep)
// MODE 1: scatter columns to outp[(n*CC+j)*HH + i]  (pass 3: final outputs)
// Always emits the bf16 packed-B layout of G_new for the next step.
template <int MODE>
__global__ __launch_bounds__(256)
void gemm_step(const unsigned short* __restrict__ Ap,
               const unsigned short* __restrict__ Bin,
               unsigned short* __restrict__ Bout,
               const float* __restrict__ wx, int j,
               float* __restrict__ outp) {
  __shared__ float slab[2][64 * 66];
  const int wg = blockIdx.x;            // M-slab of 64 rows
  const int tid = threadIdx.x;
  const int w = tid >> 5, lane = tid & 31;
  const int mtw = wg * 4 + (w & 3);     // A m-tile for this wave
  const int khalf = w >> 2;             // K split
  const int kt0 = khalf * 32;

  v8f acc[4];
  v8f z = {0.f, 0.f, 0.f, 0.f, 0.f, 0.f, 0.f, 0.f};
#pragma unroll
  for (int nt = 0; nt < 4; nt++) acc[nt] = z;

  const size_t abase = ((size_t)mtw * KT + kt0) * 512 + lane * 16;
  const unsigned short* bbase = Bin + (size_t)kt0 * 4 * 512 + lane * 16;

  v16bf a0, a1, b0[4], b1[4];
  a0 = *(const v16bf*)(Ap + abase);
#pragma unroll
  for (int nt = 0; nt < 4; nt++)
    b0[nt] = *(const v16bf*)(bbase + (size_t)nt * 512);

  // 32 K-tiles: 15 unrolled-by-2 iterations + epilogue (no buffer rotation)
  for (int i = 1; i < 31; i += 2) {
    a1 = *(const v16bf*)(Ap + abase + (size_t)i * 512);
#pragma unroll
    for (int nt = 0; nt < 4; nt++)
      b1[nt] = *(const v16bf*)(bbase + ((size_t)i * 4 + nt) * 512);
    __builtin_prefetch(Ap + abase + (size_t)(i + 2) * 512, 0, 1);
#pragma unroll
    for (int nt = 0; nt < 4; nt++)
      acc[nt] = __builtin_amdgcn_wmma_f32_16x16x32_bf16(
          false, a0, false, b0[nt], (short)0, acc[nt], false, false);

    a0 = *(const v16bf*)(Ap + abase + (size_t)(i + 1) * 512);
#pragma unroll
    for (int nt = 0; nt < 4; nt++)
      b0[nt] = *(const v16bf*)(bbase + ((size_t)(i + 1) * 4 + nt) * 512);
#pragma unroll
    for (int nt = 0; nt < 4; nt++)
      acc[nt] = __builtin_amdgcn_wmma_f32_16x16x32_bf16(
          false, a1, false, b1[nt], (short)0, acc[nt], false, false);
  }
  a1 = *(const v16bf*)(Ap + abase + (size_t)31 * 512);
#pragma unroll
  for (int nt = 0; nt < 4; nt++)
    b1[nt] = *(const v16bf*)(bbase + ((size_t)31 * 4 + nt) * 512);
#pragma unroll
  for (int nt = 0; nt < 4; nt++)
    acc[nt] = __builtin_amdgcn_wmma_f32_16x16x32_bf16(
        false, a0, false, b0[nt], (short)0, acc[nt], false, false);
#pragma unroll
  for (int nt = 0; nt < 4; nt++)
    acc[nt] = __builtin_amdgcn_wmma_f32_16x16x32_bf16(
        false, a1, false, b1[nt], (short)0, acc[nt], false, false);

  // dump partial D into this K-half's LDS slab
  const int rbase = (w & 3) * 16 + ((lane >> 4) << 3);
#pragma unroll
  for (int nt = 0; nt < 4; nt++) {
    int n = nt * 16 + (lane & 15);
#pragma unroll
    for (int r = 0; r < 8; r++) slab[khalf][(rbase + r) * 66 + n] = acc[nt][r];
  }
  __syncthreads();

  if (MODE == 0) {
    // plain f32 G: [i*NCK + n], 16 elements per thread
    int il = tid >> 2, n0 = (tid & 3) * 16;
    float4* dst = (float4*)(outp + (size_t)(wg * 64 + il) * NCK + n0);
#pragma unroll
    for (int c = 0; c < 4; c++) {
      float v0 = slab[0][il * 66 + n0 + 4 * c + 0] + slab[1][il * 66 + n0 + 4 * c + 0] + wx[(n0 + 4 * c + 0) * CC + j];
      float v1 = slab[0][il * 66 + n0 + 4 * c + 1] + slab[1][il * 66 + n0 + 4 * c + 1] + wx[(n0 + 4 * c + 1) * CC + j];
      float v2 = slab[0][il * 66 + n0 + 4 * c + 2] + slab[1][il * 66 + n0 + 4 * c + 2] + wx[(n0 + 4 * c + 2) * CC + j];
      float v3 = slab[0][il * 66 + n0 + 4 * c + 3] + slab[1][il * 66 + n0 + 4 * c + 3] + wx[(n0 + 4 * c + 3) * CC + j];
      dst[c] = make_float4(v0, v1, v2, v3);
    }
  } else {
    // output scatter: out[(n*CC+j)*HH + i], 16 elements per thread
    int nl = tid >> 2, i0 = (tid & 3) * 16;
    float wv = wx[nl * CC + j];
    float4* dst = (float4*)(outp + (size_t)(nl * CC + j) * HH + wg * 64 + i0);
#pragma unroll
    for (int c = 0; c < 4; c++) {
      float v0 = slab[0][(i0 + 4 * c + 0) * 66 + nl] + slab[1][(i0 + 4 * c + 0) * 66 + nl] + wv;
      float v1 = slab[0][(i0 + 4 * c + 1) * 66 + nl] + slab[1][(i0 + 4 * c + 1) * 66 + nl] + wv;
      float v2 = slab[0][(i0 + 4 * c + 2) * 66 + nl] + slab[1][(i0 + 4 * c + 2) * 66 + nl] + wv;
      float v3 = slab[0][(i0 + 4 * c + 3) * 66 + nl] + slab[1][(i0 + 4 * c + 3) * 66 + nl] + wv;
      dst[c] = make_float4(v0, v1, v2, v3);
    }
  }

  // re-pack G_new (= slab0+slab1+wx) into bf16 B-fragment layout for next step
  {
    int tl = tid >> 5, ln = tid & 31;           // 8 tiles x 32 lanes = 256 threads
    int ktl = tl >> 2, nt = tl & 3;
    int n = nt * 16 + (ln & 15);
    float wv = wx[n * CC + j];
    Bf16x16 wvv;
#pragma unroll
    for (int q = 0; q < 16; q++) {
      int kk = ktl * 32 + kOf(ln, q);
      float v = slab[0][kk * 66 + n] + slab[1][kk * 66 + n] + wv;
      wvv.u[q] = f2bf(v);
    }
    *(v16bf*)(Bout + (((size_t)(wg * 2 + ktl)) * 4 + nt) * 512 + ln * 16) = wvv.v;
  }
}

// ---------------- full 2048^3 bf16 GEMM (repeated squaring of U) -------------
__global__ __launch_bounds__(128)
void gemm_sq(const unsigned short* __restrict__ Ap,
             const unsigned short* __restrict__ Bp,
             float* __restrict__ Dplain) {
  __shared__ float slab[64 * 66];
  const int tid = threadIdx.x;
  const int w = tid >> 5, lane = tid & 31;
  const int mtw = blockIdx.x * 4 + w;
  const int ntg0 = blockIdx.y * 4;

  v8f acc[4];
  v8f z = {0.f, 0.f, 0.f, 0.f, 0.f, 0.f, 0.f, 0.f};
#pragma unroll
  for (int nt = 0; nt < 4; nt++) acc[nt] = z;

  const size_t abase = (size_t)mtw * KT * 512 + lane * 16;
  const unsigned short* bbase = Bp + (size_t)ntg0 * 512 + lane * 16;
  const size_t brow = (size_t)(HH / 16) * 512;     // B tile-row stride in elems

  v16bf a0, a1, b0[4], b1[4];
  a0 = *(const v16bf*)(Ap + abase);
#pragma unroll
  for (int nt = 0; nt < 4; nt++)
    b0[nt] = *(const v16bf*)(bbase + (size_t)nt * 512);

  // 64 K-tiles: 31 unrolled-by-2 iterations + epilogue
  for (int i = 1; i < 63; i += 2) {
    a1 = *(const v16bf*)(Ap + abase + (size_t)i * 512);
#pragma unroll
    for (int nt = 0; nt < 4; nt++)
      b1[nt] = *(const v16bf*)(bbase + (size_t)i * brow + (size_t)nt * 512);
    __builtin_prefetch(Ap + abase + (size_t)(i + 2) * 512, 0, 1);
#pragma unroll
    for (int nt = 0; nt < 4; nt++)
      acc[nt] = __builtin_amdgcn_wmma_f32_16x16x32_bf16(
          false, a0, false, b0[nt], (short)0, acc[nt], false, false);

    a0 = *(const v16bf*)(Ap + abase + (size_t)(i + 1) * 512);
#pragma unroll
    for (int nt = 0; nt < 4; nt++)
      b0[nt] = *(const v16bf*)(bbase + (size_t)(i + 1) * brow + (size_t)nt * 512);
#pragma unroll
    for (int nt = 0; nt < 4; nt++)
      acc[nt] = __builtin_amdgcn_wmma_f32_16x16x32_bf16(
          false, a1, false, b1[nt], (short)0, acc[nt], false, false);
  }
  a1 = *(const v16bf*)(Ap + abase + (size_t)63 * 512);
#pragma unroll
  for (int nt = 0; nt < 4; nt++)
    b1[nt] = *(const v16bf*)(bbase + (size_t)63 * brow + (size_t)nt * 512);
#pragma unroll
  for (int nt = 0; nt < 4; nt++)
    acc[nt] = __builtin_amdgcn_wmma_f32_16x16x32_bf16(
        false, a0, false, b0[nt], (short)0, acc[nt], false, false);
#pragma unroll
  for (int nt = 0; nt < 4; nt++)
    acc[nt] = __builtin_amdgcn_wmma_f32_16x16x32_bf16(
        false, a1, false, b1[nt], (short)0, acc[nt], false, false);

  const int rbase = w * 16 + ((lane >> 4) << 3);
#pragma unroll
  for (int nt = 0; nt < 4; nt++) {
    int n = nt * 16 + (lane & 15);
#pragma unroll
    for (int r = 0; r < 8; r++) slab[(rbase + r) * 66 + n] = acc[nt][r];
  }
  __syncthreads();

  int il = tid >> 1, n0 = (tid & 1) * 32;
  float4* dst = (float4*)(Dplain + (size_t)(blockIdx.x * 64 + il) * HH + blockIdx.y * 64 + n0);
#pragma unroll
  for (int c = 0; c < 8; c++) {
    dst[c] = make_float4(slab[il * 66 + n0 + 4 * c + 0], slab[il * 66 + n0 + 4 * c + 1],
                         slab[il * 66 + n0 + 4 * c + 2], slab[il * 66 + n0 + 4 * c + 3]);
  }
}

// ---------------- e0 = local end of chunk 0 ----------------
__global__ void copy_e0(const float* __restrict__ Gplain, float* __restrict__ est,
                        float* __restrict__ e0) {
  int i = blockIdx.x * blockDim.x + threadIdx.x;
  if (i >= HH) return;
  float v = Gplain[(size_t)i * NCK + 0];
  est[(size_t)i * NCK + 0] = v;
  e0[i] = v;
}

// ---------------- chunk-boundary scan: e_k = U^64 * e_{k-1} + local_k --------
// P is plain row-major bf16 (halves serial L2 traffic on this chain)
__global__ void matvec_scan(const unsigned short* __restrict__ Pbf,
                            const float* __restrict__ ein,
                            const float* __restrict__ Gplain, int k,
                            float* __restrict__ eout, float* __restrict__ est) {
  int lane = threadIdx.x & 31;
  int row  = blockIdx.x * 8 + (threadIdx.x >> 5);
  float s = 0.f;
  const unsigned short* prow = Pbf + (size_t)row * HH;
  for (int it = 0; it < HH / 128; it++) {
    int j = it * 128 + lane * 4;
    uint2 p = *(const uint2*)(prow + j);
    s += bf2f(p.x & 0xffffu) * ein[j + 0];
    s += bf2f(p.x >> 16)     * ein[j + 1];
    s += bf2f(p.y & 0xffffu) * ein[j + 2];
    s += bf2f(p.y >> 16)     * ein[j + 3];
  }
  for (int off = 16; off > 0; off >>= 1) s += __shfl_xor(s, off, 32);
  if (lane == 0) {
    float v = s + Gplain[(size_t)row * NCK + k];
    eout[row] = v;
    est[(size_t)row * NCK + k] = v;
  }
}

// ---------------- host launch ----------------
extern "C" void kernel_launch(void* const* d_in, const int* in_sizes, int n_in,
                              void* d_out, int out_size, void* d_ws, size_t ws_size,
                              hipStream_t stream) {
  (void)in_sizes; (void)n_in; (void)out_size; (void)ws_size;
  const float* inp = (const float*)d_in[0];
  const float* Wv  = (const float*)d_in[1];
  const float* U   = (const float*)d_in[2];
  const float* h0  = (const float*)d_in[3];
  float* out = (float*)d_out;

  char* ws = (char*)d_ws;
  size_t off = 0;
  float* wx = (float*)(ws + off);                 off += (size_t)TT * 4;                 // 16 KB
  float* Gplain = (float*)(ws + off);             off += (size_t)HH * NCK * 4;           // 512 KB
  float* est    = (float*)(ws + off);             off += (size_t)HH * NCK * 4;           // 512 KB
  float* evec   = (float*)(ws + off);             off += (size_t)2 * HH * 4;             // 16 KB
  float* Pplain = (float*)(ws + off);             off += (size_t)HH * HH * 4;            // 16 MB
  unsigned short* UA = (unsigned short*)(ws + off); off += (size_t)HH * HH * 2;          // 8 MB
  unsigned short* Aq = (unsigned short*)(ws + off); off += (size_t)HH * HH * 2;          // 8 MB
  unsigned short* Bq = (unsigned short*)(ws + off); off += (size_t)HH * HH * 2;          // 8 MB
  unsigned short* Pbf = (unsigned short*)(ws + off); off += (size_t)HH * HH * 2;         // 8 MB
  unsigned short* Bp[2];
  Bp[0] = (unsigned short*)(ws + off);            off += (size_t)HH * NCK * 2;           // 256 KB
  Bp[1] = (unsigned short*)(ws + off);            off += (size_t)HH * NCK * 2;           // 256 KB

  // 1) wx[t] = <inputs[t], W>
  wx_kernel<<<TT / 8, 256, 0, stream>>>(inp, Wv, wx);

  // 2) one-time packs of U
  pack_a<<<(MT * KT * 32) / 256, 256, 0, stream>>>(U, UA);
  pack_b_full<<<(KT * (HH / 16) * 32) / 256, 256, 0, stream>>>(U, Bq);

  // 3) pass 1: local chunk sweep (col0 seeded with h0, others zero)
  init_pass1_b<<<(KT * 4 * 32) / 256, 256, 0, stream>>>(h0, Bp[0]);
  for (int j = 0; j < CC; j++) {
    gemm_step<0><<<HH / 64, 256, 0, stream>>>(UA, Bp[j & 1], Bp[(j + 1) & 1], wx, j, Gplain);
  }

  // 4) U^64 by repeated squaring (6 bf16 GEMMs), then plain-bf16 copy for scan
  dim3 gsq(HH / 64, HH / 64);
  for (int s = 0; s < 6; s++) {
    gemm_sq<<<gsq, 128, 0, stream>>>(s == 0 ? UA : Aq, Bq, Pplain);
    if (s < 5) {
      pack_a<<<(MT * KT * 32) / 256, 256, 0, stream>>>(Pplain, Aq);
      pack_b_full<<<(KT * (HH / 16) * 32) / 256, 256, 0, stream>>>(Pplain, Bq);
    }
  }
  pack_plain_bf16<<<(HH * HH / 4) / 256, 256, 0, stream>>>(Pplain, Pbf);

  // 5) chunk-boundary scan (the second serial chain)
  copy_e0<<<HH / 256, 256, 0, stream>>>(Gplain, est, evec);
  for (int k = 1; k < NCK; k++) {
    matvec_scan<<<HH / 8, 256, 0, stream>>>(Pbf, evec + ((k - 1) & 1) * HH, Gplain, k,
                                            evec + (k & 1) * HH, est);
  }

  // 6) pass 3: corrected sweep, scattering every step into d_out
  init_pass3_b<<<(KT * 4 * 32) / 256, 256, 0, stream>>>(h0, est, Bp[0]);
  for (int j = 0; j < CC; j++) {
    gemm_step<1><<<HH / 64, 256, 0, stream>>>(UA, Bp[j & 1], Bp[(j + 1) & 1], wx, j, out);
  }
}
